// MultiHeadCausalAttention_8340826489270
// MI455X (gfx1250) — compile-verified, matrix operand order checked by reading, not run on hardware
//
#include <hip/hip_runtime.h>
#include <hip/hip_bf16.h>

// ---------------- CDNA5 (gfx1250) types -------------------------------------
typedef __attribute__((ext_vector_type(16))) __bf16       v16bf;
typedef __attribute__((ext_vector_type(8)))  float        v8f;
typedef __attribute__((ext_vector_type(4)))  unsigned int u32x4;
typedef __attribute__((ext_vector_type(8)))  unsigned int u32x8;

#define D_MODEL 768
#define NHEADS  12
#define HDIM    64
#define BB      2
#define TT      4096
#define MTOT    (BB * TT)      // 8192 rows (B*T)
#define NQKV    (3 * D_MODEL)  // 2304

// ---------------- helpers ----------------------------------------------------
static __device__ __forceinline__ unsigned short f32_to_bf16(float f) {
  unsigned int u = __builtin_bit_cast(unsigned int, f);
  u += 0x7FFFu + ((u >> 16) & 1u);          // round-to-nearest-even
  return (unsigned short)(u >> 16);
}

// Load one 16x32 bf16 A/B fragment from a row-major [rows x K] matrix.
// Caller passes p = base + row*K + kchunk  where row = tile_row + (lane&15)
// and kchunk = (lane>>4)*8.  Per ISA layout: lanes 0-15 hold K {0..7,16..23},
// lanes 16-31 hold K {8..15,24..31} -> two 16-byte loads per lane.
static __device__ __forceinline__ v16bf load_frag16(const unsigned short* p) {
  u32x4 lo = *reinterpret_cast<const u32x4*>(p);       // K +0..7
  u32x4 hi = *reinterpret_cast<const u32x4*>(p + 16);  // K +16..23
  u32x8 t;
  t[0] = lo[0]; t[1] = lo[1]; t[2] = lo[2]; t[3] = lo[3];
  t[4] = hi[0]; t[5] = hi[1]; t[6] = hi[2]; t[7] = hi[3];
  return __builtin_bit_cast(v16bf, t);
}

static __device__ __forceinline__ v8f wmma_bf16(v16bf a, v16bf b, v8f c) {
  return __builtin_amdgcn_wmma_f32_16x16x32_bf16(false, a, false, b,
                                                 (short)0, c, false, false);
}

// ---------------- f32 -> bf16 convert ----------------------------------------
__global__ void cvt_bf16_kernel(const float* __restrict__ in,
                                unsigned short* __restrict__ out, int n) {
  int i = blockIdx.x * blockDim.x + threadIdx.x;
  if (i < n) out[i] = f32_to_bf16(in[i]);
}

// ---------------- wave-tile GEMM core: 64(M) x 64(N) per wave ----------------
// C[m,n] = sum_k A[m,k] * W[n,k]   (A, W row-major bf16, stride K)
// 8 fragment loads -> 16 WMMAs per K-step of 32.
static __device__ __forceinline__ void gemm_acc(const unsigned short* __restrict__ A,
                                                const unsigned short* __restrict__ W,
                                                int K, int m0, int n0, int lane,
                                                v8f c[4][4]) {
  const int lnk = lane & 15;
  const int hi8 = (lane >> 4) << 3;
  for (int kk = 0; kk < K; kk += 32) {
    if (kk + 32 < K)
      __builtin_prefetch(A + (size_t)(m0 + lnk) * K + kk + 32, 0, 1);
    v16bf af[4], wf[4];
    for (int i = 0; i < 4; ++i)
      af[i] = load_frag16(A + (size_t)(m0 + i * 16 + lnk) * K + kk + hi8);
    for (int j = 0; j < 4; ++j)
      wf[j] = load_frag16(W + (size_t)(n0 + j * 16 + lnk) * K + kk + hi8);
    for (int i = 0; i < 4; ++i)
      for (int j = 0; j < 4; ++j)
        c[i][j] = wmma_bf16(af[i], wf[j], c[i][j]);
  }
}

// ---------------- kernel 1: QKV projection -----------------------------------
// qkv = x @ w_qkv^T + b_qkv ; scatter to Q[B,H,T,64]*scale, K[B,H,T,64],
// V^T[B,H,64,T] (transposed so the PV WMMA B-operand loads contiguously).
// NOTE: part / head / batch are wave-uniform (768 % 64 == 0, 4096 % 64 == 0),
// so the scatter branch is scalar and each arm is a tight loop (no spill).
__global__ __launch_bounds__(256, 1) void qkv_kernel(
    const unsigned short* __restrict__ xb, const unsigned short* __restrict__ wb,
    const float* __restrict__ bias, unsigned short* __restrict__ qb,
    unsigned short* __restrict__ kb, unsigned short* __restrict__ vtb) {
  const int lane = threadIdx.x & 31;
  const int wid  = blockIdx.x * 8 + (threadIdx.x >> 5);
  const int wn   = wid % (NQKV / 64);   // 36 column wave-tiles
  const int wm   = wid / (NQKV / 64);   // 128 row wave-tiles
  const int m0 = wm * 64, n0 = wn * 64;

  v8f c[4][4];
  for (int i = 0; i < 4; ++i)
    for (int j = 0; j < 4; ++j)
      for (int r = 0; r < 8; ++r) c[i][j][r] = 0.f;

  gemm_acc(xb, wb, D_MODEL, m0, n0, lane, c);

  const int lnk = lane & 15, hi8 = (lane >> 4) << 3;
  const int part = n0 / D_MODEL;            // 0=q 1=k 2=v (wave-uniform)
  const int cc0  = n0 % D_MODEL;            // multiple of 64 (wave-uniform)
  const int bidx = m0 >> 12;                // batch (wave-uniform)
  const int t0   = m0 & (TT - 1);           // row base within T (wave-uniform)

  if (part == 0) {                          // Q: [B,H,T,64], fold 1/sqrt(64)
    for (int j = 0; j < 4; ++j) {
      const int h = (cc0 + j * 16) >> 6;
      const int d = j * 16 + lnk;
      const float bv = bias[n0 + j * 16 + lnk];
      unsigned short* dst =
          qb + ((size_t)(bidx * NHEADS + h) * TT + t0) * HDIM + d;
      for (int i = 0; i < 4; ++i)
        for (int r = 0; r < 8; ++r)
          dst[(size_t)(i * 16 + r + hi8) * HDIM] =
              f32_to_bf16((c[i][j][r] + bv) * 0.125f);
    }
  } else if (part == 1) {                   // K: [B,H,T,64]
    for (int j = 0; j < 4; ++j) {
      const int h = (cc0 + j * 16) >> 6;
      const int d = j * 16 + lnk;
      const float bv = bias[n0 + j * 16 + lnk];
      unsigned short* dst =
          kb + ((size_t)(bidx * NHEADS + h) * TT + t0) * HDIM + d;
      for (int i = 0; i < 4; ++i)
        for (int r = 0; r < 8; ++r)
          dst[(size_t)(i * 16 + r + hi8) * HDIM] = f32_to_bf16(c[i][j][r] + bv);
    }
  } else {                                  // V^T: [B,H,64,T]
    for (int j = 0; j < 4; ++j) {
      const int h = (cc0 + j * 16) >> 6;
      const int d = j * 16 + lnk;
      const float bv = bias[n0 + j * 16 + lnk];
      unsigned short* dst =
          vtb + ((size_t)(bidx * NHEADS + h) * HDIM + d) * TT + t0;
      for (int i = 0; i < 4; ++i)
        for (int r = 0; r < 8; ++r)
          dst[i * 16 + r + hi8] = f32_to_bf16(c[i][j][r] + bv);
    }
  }
}

// ---------------- kernel 2: causal flash attention ---------------------------
// One wave per 16 q-rows per (b,h). 64 keys per iteration (4 score tiles).
__global__ __launch_bounds__(128, 1) void attn_kernel(
    const unsigned short* __restrict__ qb, const unsigned short* __restrict__ kb,
    const unsigned short* __restrict__ vtb, unsigned short* __restrict__ attnb) {
  alignas(16) __shared__ unsigned short smem[4 * 16 * 64];  // 2KB per wave
  const int lane  = threadIdx.x & 31;
  const int wslot = threadIdx.x >> 5;
  const int wid   = blockIdx.x * 4 + wslot;
  const int qt = wid & 255;                 // T/16 = 256 q-tiles
  const int bh = wid >> 8;                  // 0..23
  const int lnk = lane & 15;
  const int hi8 = (lane >> 4) << 3;
  const int qrow0 = qt << 4;

  const unsigned short* qh = qb  + (size_t)bh * TT * HDIM;
  const unsigned short* kh = kb  + (size_t)bh * TT * HDIM;
  const unsigned short* vh = vtb + (size_t)bh * HDIM * TT; // [64][T]
  unsigned short* lw = smem + wslot * (16 * 64);

  v16bf aq[2];
  for (int kt = 0; kt < 2; ++kt)
    aq[kt] = load_frag16(qh + (size_t)(qrow0 + lnk) * HDIM + kt * 32 + hi8);

  v8f o[4];
  float m_i[8], l_i[8];
  for (int f = 0; f < 4; ++f)
    for (int r = 0; r < 8; ++r) o[f][r] = 0.f;
  for (int r = 0; r < 8; ++r) { m_i[r] = -1e30f; l_i[r] = 0.f; }

  const int nit = (qt >> 2) + 1;            // 64-key blocks to reach diagonal
  for (int it = 0; it < nit; ++it) {
    const int kv0 = it << 6;

    // ---- S = (Q*scale) @ K^T : four 16x16 tiles, K-dim 64 = 2 wmma steps
    v8f s[4];
    for (int jc = 0; jc < 4; ++jc)
      for (int r = 0; r < 8; ++r) s[jc][r] = 0.f;
    for (int jc = 0; jc < 4; ++jc)
      for (int kt = 0; kt < 2; ++kt) {
        v16bf bk = load_frag16(kh + (size_t)(kv0 + jc * 16 + lnk) * HDIM +
                               kt * 32 + hi8);
        s[jc] = wmma_bf16(aq[kt], bk, s[jc]);
      }

    // ---- causal mask (only the diagonal iteration can cross it)
    if (it == nit - 1) {
      for (int jc = 0; jc < 4; ++jc)
        for (int r = 0; r < 8; ++r) {
          const int row = qrow0 + r + hi8;
          const int col = kv0 + jc * 16 + lnk;
          if (col > row) s[jc][r] = -1e30f;
        }
    }

    // ---- online softmax (row reduction across 16-lane halves)
    for (int r = 0; r < 8; ++r) {
      float v = fmaxf(fmaxf(s[0][r], s[1][r]), fmaxf(s[2][r], s[3][r]));
      v = fmaxf(v, __shfl_xor(v, 1, 32));
      v = fmaxf(v, __shfl_xor(v, 2, 32));
      v = fmaxf(v, __shfl_xor(v, 4, 32));
      v = fmaxf(v, __shfl_xor(v, 8, 32));
      const float mn = fmaxf(m_i[r], v);
      const float al = __expf(m_i[r] - mn);
      m_i[r] = mn;
      float rs = 0.f;
      for (int jc = 0; jc < 4; ++jc) {
        const float p = __expf(s[jc][r] - mn);
        s[jc][r] = p;
        rs += p;
      }
      rs += __shfl_xor(rs, 1, 32);
      rs += __shfl_xor(rs, 2, 32);
      rs += __shfl_xor(rs, 4, 32);
      rs += __shfl_xor(rs, 8, 32);
      l_i[r] = l_i[r] * al + rs;
      for (int f = 0; f < 4; ++f) o[f][r] *= al;
    }

    // ---- re-layout P (C-frag) -> A-frag via per-wave LDS (16x64 bf16)
    for (int jc = 0; jc < 4; ++jc)
      for (int r = 0; r < 8; ++r)
        lw[(r + hi8) * 64 + jc * 16 + lnk] = f32_to_bf16(s[jc][r]);
    asm volatile("s_wait_dscnt 0" ::: "memory");   // same-wave LDS ordering
    v16bf pa[2];
    for (int kt = 0; kt < 2; ++kt)
      pa[kt] = load_frag16(lw + lnk * 64 + kt * 32 + hi8);

    // ---- O += P @ V  (V^T rows are the dims -> contiguous in t)
    for (int kt = 0; kt < 2; ++kt)
      for (int f = 0; f < 4; ++f) {
        v16bf bv = load_frag16(vh + (size_t)(f * 16 + lnk) * TT + kv0 +
                               kt * 32 + hi8);
        o[f] = wmma_bf16(pa[kt], bv, o[f]);
      }
  }

  // ---- normalize and store [B,T,C] with heads re-interleaved
  const int b = bh / NHEADS, h = bh % NHEADS;
  for (int r = 0; r < 8; ++r) {
    const float inv = 1.0f / l_i[r];
    const int row = qrow0 + r + hi8;
    const size_t base = (size_t)(b * TT + row) * D_MODEL + h * HDIM;
    for (int f = 0; f < 4; ++f)
      attnb[base + f * 16 + lnk] = f32_to_bf16(o[f][r] * inv);
  }
}

// ---------------- kernel 3: output projection --------------------------------
__global__ __launch_bounds__(256, 1) void out_kernel(
    const unsigned short* __restrict__ ab, const unsigned short* __restrict__ wb,
    const float* __restrict__ bias, float* __restrict__ out) {
  const int lane = threadIdx.x & 31;
  const int wid  = blockIdx.x * 8 + (threadIdx.x >> 5);
  const int wn   = wid % (D_MODEL / 64);  // 12
  const int wm   = wid / (D_MODEL / 64);  // 128
  const int m0 = wm * 64, n0 = wn * 64;

  v8f c[4][4];
  for (int i = 0; i < 4; ++i)
    for (int j = 0; j < 4; ++j)
      for (int r = 0; r < 8; ++r) c[i][j][r] = 0.f;

  gemm_acc(ab, wb, D_MODEL, m0, n0, lane, c);

  const int lnk = lane & 15, hi8 = (lane >> 4) << 3;
  for (int j = 0; j < 4; ++j) {
    const int col  = n0 + j * 16 + lnk;
    const float bv = bias[col];
    float* dst = out + (size_t)m0 * D_MODEL + col;
    for (int i = 0; i < 4; ++i)
      for (int r = 0; r < 8; ++r)
        dst[(size_t)(i * 16 + r + hi8) * D_MODEL] = c[i][j][r] + bv;
  }
}

// ---------------- host launcher ----------------------------------------------
extern "C" void kernel_launch(void* const* d_in, const int* in_sizes, int n_in,
                              void* d_out, int out_size, void* d_ws, size_t ws_size,
                              hipStream_t stream) {
  const float* x     = (const float*)d_in[0];
  const float* w_qkv = (const float*)d_in[1];
  const float* b_qkv = (const float*)d_in[2];
  const float* w_out = (const float*)d_in[3];
  const float* b_out = (const float*)d_in[4];
  float* out = (float*)d_out;

  // workspace carve-up (all chunks 256B aligned by construction)
  char* ws = (char*)d_ws;
  unsigned short* xb    = (unsigned short*)ws; ws += (size_t)MTOT * D_MODEL * 2;
  unsigned short* wqkvb = (unsigned short*)ws; ws += (size_t)NQKV * D_MODEL * 2;
  unsigned short* woutb = (unsigned short*)ws; ws += (size_t)D_MODEL * D_MODEL * 2;
  unsigned short* qb    = (unsigned short*)ws; ws += (size_t)MTOT * D_MODEL * 2;
  unsigned short* kb    = (unsigned short*)ws; ws += (size_t)MTOT * D_MODEL * 2;
  unsigned short* vtb   = (unsigned short*)ws; ws += (size_t)MTOT * D_MODEL * 2;
  unsigned short* attnb = (unsigned short*)ws;
  (void)in_sizes; (void)n_in; (void)out_size; (void)ws_size;

  const int nx = MTOT * D_MODEL;      // 6,291,456
  const int nw = NQKV * D_MODEL;      // 1,769,472
  const int no = D_MODEL * D_MODEL;   //   589,824
  cvt_bf16_kernel<<<(nx + 255) / 256, 256, 0, stream>>>(x, xb, nx);
  cvt_bf16_kernel<<<(nw + 255) / 256, 256, 0, stream>>>(w_qkv, wqkvb, nw);
  cvt_bf16_kernel<<<(no + 255) / 256, 256, 0, stream>>>(w_out, woutb, no);

  // QKV GEMM: 128 M-wavetiles x 36 N-wavetiles = 4608 waves / 8 per block
  qkv_kernel<<<(128 * 36) / 8, 256, 0, stream>>>(xb, wqkvb, b_qkv, qb, kb, vtb);

  // attention: B*H*(T/16) = 6144 waves, 4 waves per block
  attn_kernel<<<(BB * NHEADS * (TT / 16)) / 4, 128, 0, stream>>>(qb, kb, vtb, attnb);

  // output GEMM: 128 x 12 = 1536 waves / 8 per block
  out_kernel<<<(128 * 12) / 8, 256, 0, stream>>>(attnb, woutb, b_out, out);
}
